// GNN_17652315586927
// MI455X (gfx1250) — compile-verified, matrix operand order checked by reading, not run on hardware
//
#include <hip/hip_runtime.h>
#include <hip/hip_bf16.h>

#define NL 8192
#define NA 4096
#define NG 256
#define HD 256
#define OD 128

typedef float v8f __attribute__((ext_vector_type(8)));
typedef __bf16 bf16x16 __attribute__((ext_vector_type(16)));
typedef __bf16 bf16x8  __attribute__((ext_vector_type(8)));

union FragU { bf16x16 v; struct { bf16x8 lo, hi; } p; };

__device__ __forceinline__ unsigned short f2bf(float f) {
  unsigned int u = __float_as_uint(f);
  u += 0x7fffu + ((u >> 16) & 1u);            // round-to-nearest-even
  return (unsigned short)(u >> 16);
}

// ---------------- elementwise / scatter kernels ----------------

__global__ void k_zero(float* __restrict__ p, int n) {
  int i = blockIdx.x * blockDim.x + threadIdx.x;
  if (i < n) p[i] = 0.f;
}

__global__ void k_count(const int* __restrict__ dst, float* __restrict__ cnt, int nE) {
  int e = blockIdx.x * blockDim.x + threadIdx.x;
  if (e < nE) atomicAdd(&cnt[dst[e]], 1.0f);
}

// 64 lanes per edge, float4 per lane: acc[dst] += srcMat[src] (256-float rows)
__global__ void k_scatter(const float* __restrict__ srcMat, const int* __restrict__ src,
                          const int* __restrict__ dst, float* __restrict__ acc, int nE) {
  int t = blockIdx.x * blockDim.x + threadIdx.x;
  int e = t >> 6;
  if (e >= nE) return;
  int c4 = (t & 63) << 2;
  int s = src[e], d = dst[e];
  const float4 v = *(const float4*)(srcMat + (size_t)s * HD + c4);
  float* o = acc + (size_t)d * HD + c4;
  atomicAdd(o + 0, v.x);
  atomicAdd(o + 1, v.y);
  atomicAdd(o + 2, v.z);
  atomicAdd(o + 3, v.w);
}

// h = relu(agg/max(cnt,1) + Wr + b)   (single relation)
__global__ void k_hfin1(float* __restrict__ h, const float* __restrict__ agg,
                        const float* __restrict__ cnt, const float* __restrict__ Wr,
                        const float* __restrict__ b, int total) {
  int i = blockIdx.x * blockDim.x + threadIdx.x;
  if (i >= total) return;
  int row = i / HD, col = i & (HD - 1);
  float v = agg[i] / fmaxf(cnt[row], 1.f) + Wr[i] + b[col];
  h[i] = fmaxf(v, 0.f);
}

// h = relu(0.5*((agg1/c1 + Wr1 + b1) + (agg2/c2 + Wr2 + b2)))   (livro)
__global__ void k_hfin2(float* __restrict__ h,
                        const float* __restrict__ agg1, const float* __restrict__ cnt1,
                        const float* __restrict__ Wr1, const float* __restrict__ b1v,
                        const float* __restrict__ agg2, const float* __restrict__ cnt2,
                        const float* __restrict__ Wr2, const float* __restrict__ b2v,
                        int total) {
  int i = blockIdx.x * blockDim.x + threadIdx.x;
  if (i >= total) return;
  int row = i / HD, col = i & (HD - 1);
  float v1 = agg1[i] / fmaxf(cnt1[row], 1.f) + Wr1[i] + b1v[col];
  float v2 = agg2[i] / fmaxf(cnt2[row], 1.f) + Wr2[i] + b2v[col];
  h[i] = fmaxf(0.5f * (v1 + v2), 0.f);
}

// X(n x K) bf16 = [agg1/cnt1 | (agg2/cnt2)? | h]
__global__ void k_build_x(unsigned short* __restrict__ X, int K,
                          const float* __restrict__ agg1, const float* __restrict__ cnt1,
                          const float* __restrict__ agg2, const float* __restrict__ cnt2,
                          const float* __restrict__ h, int total) {
  int i = blockIdx.x * blockDim.x + threadIdx.x;
  if (i >= total) return;
  int row = i / K, col = i % K;
  float v;
  if (col < HD) {
    v = agg1[(size_t)row * HD + col] / fmaxf(cnt1[row], 1.f);
  } else if (agg2 && col < 2 * HD) {
    v = agg2[(size_t)row * HD + (col - HD)] / fmaxf(cnt2[row], 1.f);
  } else {
    int hc = col - (agg2 ? 2 * HD : HD);
    v = h[(size_t)row * HD + hc];
  }
  X[i] = f2bf(v);
}

// WT(OD x K) bf16, N-major transpose of [blk0 ; blk1 ; (blk2 (+blk2b))], blocks are (HD x OD)
__global__ void k_build_wT(unsigned short* __restrict__ WT, int K,
                           const float* __restrict__ b0, const float* __restrict__ b1,
                           const float* __restrict__ b2, const float* __restrict__ b2b,
                           int total) {
  int i = blockIdx.x * blockDim.x + threadIdx.x;
  if (i >= total) return;
  int n = i / K, k = i % K;
  int blk = k / HD, kk = k & (HD - 1);
  const float* p = (blk == 0) ? b0 : ((blk == 1) ? b1 : b2);
  float v = p[kk * OD + n];
  if (blk == 2 && b2b) v += b2b[kk * OD + n];
  WT[i] = f2bf(v);
}

__global__ void k_build_bias(const float* __restrict__ b2, float* __restrict__ bl,
                             float* __restrict__ ba, float* __restrict__ bg) {
  int t = threadIdx.x;
  if (t < OD) {
    ba[t] = b2[0 * OD + t];
    bg[t] = b2[2 * OD + t];
    bl[t] = b2[1 * OD + t] + b2[3 * OD + t];   // 0.5 drops out under L2-norm
  }
}

// ---------------- WMMA GEMM + row L2-normalize ----------------
// block = 256 threads = 8 waves; each wave: one 16x16 tile; block: 16 rows x 128 cols.
// A fragment (16-bit 16x32): lane half h, elem e -> K = (e>>3)*16 + h*8 + (e&7)
// -> two contiguous bf16x8 loads. WT is N-major so B uses the same pattern.
__global__ __launch_bounds__(256)
void k_gemm_l2norm(const unsigned short* __restrict__ X, const unsigned short* __restrict__ WT,
                   const float* __restrict__ bias, float* __restrict__ out, int K) {
  __shared__ float tile[16][OD];
  __shared__ float rinv[16];
  const int wave = threadIdx.x >> 5;
  const int lane = threadIdx.x & 31;
  const int half = lane >> 4;
  const int l15  = lane & 15;
  const int rowBase = blockIdx.x * 16;
  const int aRow = rowBase + l15;          // A: M striped over lanes 0-15 (dup 16-31)
  const int bCol = wave * 16 + l15;        // B/D: N striped over lanes

  v8f acc = {0.f, 0.f, 0.f, 0.f, 0.f, 0.f, 0.f, 0.f};
  const unsigned short* aBase = X  + (size_t)aRow * K + half * 8;
  const unsigned short* bBase = WT + (size_t)bCol * K + half * 8;
  for (int kb = 0; kb < K; kb += 32) {
    FragU a, b;
    a.p.lo = *(const bf16x8*)(aBase + kb);
    a.p.hi = *(const bf16x8*)(aBase + kb + 16);
    b.p.lo = *(const bf16x8*)(bBase + kb);
    b.p.hi = *(const bf16x8*)(bBase + kb + 16);
    acc = __builtin_amdgcn_wmma_f32_16x16x32_bf16(false, a.v, false, b.v,
                                                  (short)0, acc, false, false);
  }
  // D layout: VGPR r -> M = r + 8*half, N = l15 (+16w)
  #pragma unroll
  for (int r = 0; r < 8; ++r)
    tile[half * 8 + r][bCol] = acc[r] + bias[bCol];
  __syncthreads();
  if (threadIdx.x < 16) {
    float s = 0.f;
    for (int c = 0; c < OD; ++c) { float v = tile[threadIdx.x][c]; s += v * v; }
    rinv[threadIdx.x] = 1.0f / fmaxf(sqrtf(s), 1e-12f);
  }
  __syncthreads();
  for (int i = threadIdx.x; i < 16 * OD; i += 256) {
    int r = i >> 7, c = i & (OD - 1);
    out[(size_t)(rowBase + r) * OD + c] = tile[r][c] * rinv[r];
  }
}

// ---------------- host ----------------

extern "C" void kernel_launch(void* const* d_in, const int* in_sizes, int n_in,
                              void* d_out, int out_size, void* d_ws, size_t ws_size,
                              hipStream_t stream) {
  (void)n_in; (void)out_size; (void)ws_size;
  const float* W1_la_l = (const float*)d_in[0];
  const float* W1_la_r = (const float*)d_in[1];
  const float* b1_la   = (const float*)d_in[2];
  const float* W1_al_l = (const float*)d_in[3];
  const float* W1_al_r = (const float*)d_in[4];
  const float* b1_al   = (const float*)d_in[5];
  const float* W1_lg_l = (const float*)d_in[6];
  const float* W1_lg_r = (const float*)d_in[7];
  const float* b1_lg   = (const float*)d_in[8];
  const float* W1_gl_l = (const float*)d_in[9];
  const float* W1_gl_r = (const float*)d_in[10];
  const float* b1_gl   = (const float*)d_in[11];
  const float* W2_l    = (const float*)d_in[12];  // (4,256,128)
  const float* W2_r    = (const float*)d_in[13];
  const float* b2      = (const float*)d_in[14];
  const int* src_la = (const int*)d_in[15];
  const int* dst_la = (const int*)d_in[16];
  const int* src_al = (const int*)d_in[17];
  const int* dst_al = (const int*)d_in[18];
  const int* src_lg = (const int*)d_in[19];
  const int* dst_lg = (const int*)d_in[20];
  const int* src_gl = (const int*)d_in[21];
  const int* dst_gl = (const int*)d_in[22];
  const int E_la = in_sizes[15], E_al = in_sizes[17];
  const int E_lg = in_sizes[19], E_gl = in_sizes[21];

  // workspace layout (f32 region then bf16/u16 region; all 16B-aligned)
  float* ws    = (float*)d_ws;
  float* aggA  = ws;                              // NA*HD   (dst=autor,  rel la)
  float* aggLa = aggA  + (size_t)NA * HD;         // NL*HD   (dst=livro,  rel al)
  float* aggG  = aggLa + (size_t)NL * HD;         // NG*HD   (dst=genero, rel lg)
  float* aggLg = aggG  + (size_t)NG * HD;         // NL*HD   (dst=livro,  rel gl)
  float* cntA  = aggLg + (size_t)NL * HD;
  float* cntLa = cntA  + NA;
  float* cntG  = cntLa + NL;
  float* cntLg = cntG  + NG;
  float* h_l   = cntLg + NL;
  float* h_a   = h_l + (size_t)NL * HD;
  float* h_g   = h_a + (size_t)NA * HD;
  float* biasL = h_g + (size_t)NG * HD;
  float* biasA = biasL + OD;
  float* biasG = biasA + OD;
  unsigned short* Xl  = (unsigned short*)(biasG + OD);   // NL x 768
  unsigned short* Xa  = Xl  + (size_t)NL * 768;          // NA x 512
  unsigned short* Xg  = Xa  + (size_t)NA * 512;          // NG x 512
  unsigned short* WTl = Xg  + (size_t)NG * 512;          // 128 x 768
  unsigned short* WTa = WTl + (size_t)OD * 768;          // 128 x 512
  unsigned short* WTg = WTa + (size_t)OD * 512;          // 128 x 512

  const int B = 256;
  auto G = [](long long n) -> unsigned { return (unsigned)((n + 255) / 256); };
  const int nAgg  = NA * HD + 2 * NL * HD + NG * HD;
  const int nZero = nAgg + NA + 2 * NL + NG;

  // 1) zero aggregators + degree counters
  k_zero<<<G(nZero), B, 0, stream>>>(aggA, nZero);
  // 2) degrees (identical for both layers)
  k_count<<<G(E_la), B, 0, stream>>>(dst_la, cntA,  E_la);
  k_count<<<G(E_al), B, 0, stream>>>(dst_al, cntLa, E_al);
  k_count<<<G(E_lg), B, 0, stream>>>(dst_lg, cntG,  E_lg);
  k_count<<<G(E_gl), B, 0, stream>>>(dst_gl, cntLg, E_gl);
  // 3) layer-1 scatter of W1_*_l rows
  k_scatter<<<G((long long)E_la * 64), B, 0, stream>>>(W1_la_l, src_la, dst_la, aggA,  E_la);
  k_scatter<<<G((long long)E_al * 64), B, 0, stream>>>(W1_al_l, src_al, dst_al, aggLa, E_al);
  k_scatter<<<G((long long)E_lg * 64), B, 0, stream>>>(W1_lg_l, src_lg, dst_lg, aggG,  E_lg);
  k_scatter<<<G((long long)E_gl * 64), B, 0, stream>>>(W1_gl_l, src_gl, dst_gl, aggLg, E_gl);
  // 4) finalize hidden features (relu)
  k_hfin1<<<G(NA * HD), B, 0, stream>>>(h_a, aggA, cntA, W1_la_r, b1_la, NA * HD);
  k_hfin1<<<G(NG * HD), B, 0, stream>>>(h_g, aggG, cntG, W1_lg_r, b1_lg, NG * HD);
  k_hfin2<<<G(NL * HD), B, 0, stream>>>(h_l, aggLa, cntLa, W1_al_r, b1_al,
                                        aggLg, cntLg, W1_gl_r, b1_gl, NL * HD);
  // 5) reuse aggregators for layer 2
  k_zero<<<G(nAgg), B, 0, stream>>>(aggA, nAgg);
  // 6) layer-2 scatter of hidden rows
  k_scatter<<<G((long long)E_la * 64), B, 0, stream>>>(h_l, src_la, dst_la, aggA,  E_la);
  k_scatter<<<G((long long)E_al * 64), B, 0, stream>>>(h_a, src_al, dst_al, aggLa, E_al);
  k_scatter<<<G((long long)E_lg * 64), B, 0, stream>>>(h_l, src_lg, dst_lg, aggG,  E_lg);
  k_scatter<<<G((long long)E_gl * 64), B, 0, stream>>>(h_g, src_gl, dst_gl, aggLg, E_gl);
  // 7) build bf16 GEMM operands
  k_build_x<<<G((long long)NA * 512), B, 0, stream>>>(Xa, 512, aggA,  cntA,  nullptr, nullptr, h_a, NA * 512);
  k_build_x<<<G((long long)NG * 512), B, 0, stream>>>(Xg, 512, aggG,  cntG,  nullptr, nullptr, h_g, NG * 512);
  k_build_x<<<G((long long)NL * 768), B, 0, stream>>>(Xl, 768, aggLa, cntLa, aggLg,  cntLg,  h_l, NL * 768);
  k_build_wT<<<G(OD * 512), B, 0, stream>>>(WTa, 512, W2_l + 0 * HD * OD, W2_r + 0 * HD * OD, nullptr, nullptr, OD * 512);
  k_build_wT<<<G(OD * 512), B, 0, stream>>>(WTg, 512, W2_l + 2 * HD * OD, W2_r + 2 * HD * OD, nullptr, nullptr, OD * 512);
  k_build_wT<<<G(OD * 768), B, 0, stream>>>(WTl, 768, W2_l + 1 * HD * OD, W2_l + 3 * HD * OD,
                                            W2_r + 1 * HD * OD, W2_r + 3 * HD * OD, OD * 768);
  k_build_bias<<<1, 128, 0, stream>>>(b2, biasL, biasA, biasG);
  // 8) WMMA GEMM + L2 normalize, write [livro; autor; genero]
  float* out = (float*)d_out;
  k_gemm_l2norm<<<NL / 16, B, 0, stream>>>(Xl, WTl, biasL, out,                         768);
  k_gemm_l2norm<<<NA / 16, B, 0, stream>>>(Xa, WTa, biasA, out + (size_t)NL * OD,       512);
  k_gemm_l2norm<<<NG / 16, B, 0, stream>>>(Xg, WTg, biasG, out + (size_t)(NL + NA) * OD, 512);
}
